// LinearAttentionBlock_39487929319898
// MI455X (gfx1250) — matrix operand order repaired and checked by reading, not compiled, not run on hardware
//
#include <hip/hip_runtime.h>
#include <hip/hip_bf16.h>

// ---------------------------------------------------------------------------
// Types for CDNA5 WMMA (wave32). f16 inputs, f32 accumulate.
// ---------------------------------------------------------------------------
typedef __attribute__((ext_vector_type(16))) _Float16 v16h;
typedef __attribute__((ext_vector_type(8)))  _Float16 v8h;
typedef __attribute__((ext_vector_type(8)))  float    v8f;

#define RMS_EPS 1e-6f

// ---------------------------------------------------------------------------
// f32 -> f16 convert (flat)
// ---------------------------------------------------------------------------
__global__ void cvt_f16_kernel(const float* __restrict__ in,
                               _Float16* __restrict__ out, long n) {
  long i = (long)blockIdx.x * blockDim.x + threadIdx.x;
  if (i < n) out[i] = (_Float16)in[i];
}

// ---------------------------------------------------------------------------
// f32 [K,N] -> f16 [N,K] transposed convert (weights, done once).
// 32x32 LDS tile, 32x8 threads, both global accesses coalesced.
// ---------------------------------------------------------------------------
__global__ __launch_bounds__(256) void cvt_tr_kernel(
    const float* __restrict__ in, _Float16* __restrict__ out, int K, int N) {
  __shared__ float t[32][33];
  const int n0 = blockIdx.x * 32, k0 = blockIdx.y * 32;
  const int tx = threadIdx.x, ty = threadIdx.y;  // 32 x 8
  #pragma unroll
  for (int i = 0; i < 4; ++i)
    t[ty + 8 * i][tx] = in[(long)(k0 + ty + 8 * i) * N + n0 + tx];
  __syncthreads();
  #pragma unroll
  for (int i = 0; i < 4; ++i)
    out[(long)(n0 + ty + 8 * i) * K + k0 + tx] = (_Float16)t[tx][ty + 8 * i];
}

// ---------------------------------------------------------------------------
// Row-wise RMSNorm: out_f16[row, c] = in[row,c] * rsqrt(mean(in^2)+eps) * w[c]
// ---------------------------------------------------------------------------
__global__ __launch_bounds__(256) void rmsnorm_kernel(
    const float* __restrict__ x, const float* __restrict__ w,
    _Float16* __restrict__ out, int C) {
  const long row = blockIdx.x;
  const float* xr = x + row * (long)C;
  float ss = 0.f;
  for (int c = threadIdx.x; c < C; c += 256) { float v = xr[c]; ss = fmaf(v, v, ss); }
  for (int o = 16; o > 0; o >>= 1) ss += __shfl_xor(ss, o, 32);
  __shared__ float red[8];
  __shared__ float stot;
  if ((threadIdx.x & 31) == 0) red[threadIdx.x >> 5] = ss;
  __syncthreads();
  if (threadIdx.x == 0) {
    float t = 0.f;
    #pragma unroll
    for (int i = 0; i < 8; ++i) t += red[i];
    stot = rsqrtf(t / (float)C + RMS_EPS);
  }
  __syncthreads();
  const float s = stot;
  for (int c = threadIdx.x; c < C; c += 256)
    out[row * (long)C + c] = (_Float16)(xr[c] * s * w[c]);
}

// ---------------------------------------------------------------------------
// f16 WMMA GEMM: C[M,N] = A[M,K] * Bt[N,K]^T, fp32 accumulate.
// Bt is the PRE-TRANSPOSED weight ([N,K] row-major) so both LDS stages are
// contiguous b128 stores. M,N multiples of 128; K multiple of 32.
// 256 threads = 8 waves; block tile 128x128; wave tile 32x64 (2x4 WMMA);
// K-step 32; double-buffered LDS; global_prefetch 2 K-steps ahead.
// Fragment layouts per CDNA5 ISA:
//   A 16-bit 16x32: lane m -> K {0..7,16..23}; lane m+16 -> K {8..15,24..31}
//   B 16-bit 32x16: lane n -> K {0..7,16..23}; lane n+16 -> K {8..15,24..31}  (LDS is [n][k])
//   C f32 16x16: VGPR j -> M j (lanes 0-15) / 8+j (lanes 16-31); N = lane&15
// ---------------------------------------------------------------------------
__global__ __launch_bounds__(256) void gemm_f16_wmma_kernel(
    const _Float16* __restrict__ A, const _Float16* __restrict__ Bt,
    float* __restrict__ C, int M, int N, int K) {
  __shared__ _Float16 lA[2][128][32];
  __shared__ _Float16 lB[2][128][32];  // [n][k]

  const int tid  = threadIdx.x;
  const int lane = tid & 31;
  const int wave = tid >> 5;
  const int wm = (wave & 3) * 32;
  const int wn = (wave >> 2) * 64;
  const long bm = (long)blockIdx.y * 128;
  const long bn = (long)blockIdx.x * 128;

  v8f acc[2][4] = {};

  auto loadTiles = [&](int buf, int k0) {
    #pragma unroll
    for (int i = 0; i < 2; ++i) {
      int c = tid + 256 * i;
      int r = c >> 2, kc = (c & 3) * 8;
      const _Float16* pa = A + (bm + r) * (long)K + k0 + kc;
      const _Float16* pb = Bt + (bn + r) * (long)K + k0 + kc;
      if (k0 + 64 < K) {  // prefetch 2 K-steps ahead -> global_prefetch_b8
        __builtin_prefetch(pa + 64, 0, 1);
        __builtin_prefetch(pb + 64, 0, 1);
      }
      v8h da = *(const v8h*)pa;
      v8h db = *(const v8h*)pb;
      *(v8h*)&lA[buf][r][kc] = da;
      *(v8h*)&lB[buf][r][kc] = db;
    }
  };

  loadTiles(0, 0);
  __syncthreads();

  const int nK = K >> 5;
  int buf = 0;
  for (int kt = 0; kt < nK; ++kt) {
    if (kt + 1 < nK) loadTiles(buf ^ 1, (kt + 1) << 5);

    const int rA = lane & 15;
    const int kh = (lane >> 4) << 3;   // A: 0 or 8
    const int kb = (lane >> 4) << 4;   // unused: 0 or 16
    v16h a[2], b[4];
    #pragma unroll
    for (int mt = 0; mt < 2; ++mt) {
      v8h lo = *(const v8h*)&lA[buf][wm + mt * 16 + rA][kh];
      v8h hi = *(const v8h*)&lA[buf][wm + mt * 16 + rA][kh + 16];
      a[mt] = __builtin_shufflevector(lo, hi, 0, 1, 2, 3, 4, 5, 6, 7, 8, 9,
                                      10, 11, 12, 13, 14, 15);
    }
    #pragma unroll
    for (int nt = 0; nt < 4; ++nt) {
      v8h lo = *(const v8h*)&lB[buf][wn + nt * 16 + rA][kh];
      v8h hi = *(const v8h*)&lB[buf][wn + nt * 16 + rA][kh + 16];
      b[nt] = __builtin_shufflevector(lo, hi, 0, 1, 2, 3, 4, 5, 6, 7, 8, 9,
                                      10, 11, 12, 13, 14, 15);
    }
    #pragma unroll
    for (int mt = 0; mt < 2; ++mt)
      #pragma unroll
      for (int nt = 0; nt < 4; ++nt)
        acc[mt][nt] = __builtin_amdgcn_wmma_f32_16x16x32_f16(
            false, a[mt], false, b[nt], (short)0, acc[mt][nt], false, false);

    __syncthreads();
    buf ^= 1;
  }

  const int rlo = lane & 15;
  const int rhi = (lane >> 4) << 3;
  #pragma unroll
  for (int mt = 0; mt < 2; ++mt)
    #pragma unroll
    for (int nt = 0; nt < 4; ++nt) {
      long row0 = bm + wm + mt * 16 + rhi;
      long col  = bn + wn + nt * 16 + rlo;
      #pragma unroll
      for (int j = 0; j < 8; ++j)
        C[(row0 + j) * (long)N + col] = acc[mt][nt][j];
    }
}

// ---------------------------------------------------------------------------
// beta/g head projections (N=8 each): one wave per row, shuffle-reduce.
// beta = sigmoid(h @ wb) * mask ; g = sigmoid(h @ wg)
// ---------------------------------------------------------------------------
__global__ __launch_bounds__(32) void betag_kernel(
    const _Float16* __restrict__ hn, const float* __restrict__ wb,
    const float* __restrict__ wg, const float* __restrict__ mask,
    float* __restrict__ beta, float* __restrict__ g) {
  const long row = blockIdx.x;
  const int lane = threadIdx.x;
  float ab[8] = {}, ag[8] = {};
  for (int k = lane; k < 1024; k += 32) {
    float h = (float)hn[row * 1024 + k];
    #pragma unroll
    for (int j = 0; j < 8; ++j) {
      ab[j] = fmaf(h, wb[k * 8 + j], ab[j]);
      ag[j] = fmaf(h, wg[k * 8 + j], ag[j]);
    }
  }
  #pragma unroll
  for (int j = 0; j < 8; ++j)
    for (int o = 16; o > 0; o >>= 1) {
      ab[j] += __shfl_xor(ab[j], o, 32);
      ag[j] += __shfl_xor(ag[j], o, 32);
    }
  if (lane == 0) {
    float m = mask[row];
    #pragma unroll
    for (int j = 0; j < 8; ++j) {
      beta[row * 8 + j] = m / (1.f + expf(-ab[j]));
      g[row * 8 + j]    = 1.f / (1.f + expf(-ag[j]));
    }
  }
}

// ---------------------------------------------------------------------------
// In fused qkv [R,1536] (q:0, k:512, v:1024): L2-normalize k head rows,
// scale q by D^-0.5. One wave per (row,h); lane covers d and d+32.
// ---------------------------------------------------------------------------
__global__ __launch_bounds__(256) void prep_qk_kernel(float* __restrict__ qkv) {
  const int gw = blockIdx.x * 8 + (threadIdx.x >> 5);  // (row*8 + h)
  const int lane = threadIdx.x & 31;
  const long base = (long)(gw >> 3) * 1536 + (gw & 7) * 64;
  float* q = qkv + base;
  float* k = qkv + base + 512;
  float k0 = k[lane], k1 = k[lane + 32];
  float ss = fmaf(k0, k0, k1 * k1);
  for (int o = 16; o > 0; o >>= 1) ss += __shfl_xor(ss, o, 32);
  float inv = 1.f / (sqrtf(ss) + 1e-6f);
  k[lane]      = k0 * inv;
  k[lane + 32] = k1 * inv;
  q[lane]      *= 0.125f;   // 64^-0.5
  q[lane + 32] *= 0.125f;
}

// ---------------------------------------------------------------------------
// Delta-rule scan over fused qkv. One 64-thread block per (b,h); thread dv
// owns state column S[:,dv] in 64 VGPRs; k/q broadcast via LDS; next step's
// inputs prefetched to overlap global latency with the ~192 FMAs.
// ---------------------------------------------------------------------------
__global__ __launch_bounds__(64) void delta_scan_kernel(
    const float* __restrict__ qkv, const float* __restrict__ beta,
    const float* __restrict__ g, float* __restrict__ o) {
  const int b = blockIdx.x >> 3, h = blockIdx.x & 7;
  const int dv = threadIdx.x;
  float S[64];
  #pragma unroll
  for (int i = 0; i < 64; ++i) S[i] = 0.f;

  __shared__ float ks[64], qs[64], sc[2];
  const long base0 = (long)b * 4096 * 1536 + h * 64;  // q base; k +512; v +1024
  const long ob0   = (long)b * 4096 * 512 + h * 64;
  const long sb0   = (long)b * 4096 * 8 + h;

  float qn = qkv[base0 + dv];
  float kn = qkv[base0 + 512 + dv];
  float vn = qkv[base0 + 1024 + dv];
  float bnx = beta[sb0], gnx = g[sb0];

  for (int s = 0; s < 4096; ++s) {
    __syncthreads();
    ks[dv] = kn; qs[dv] = qn;
    if (dv == 0) { sc[0] = bnx; sc[1] = gnx; }
    float vv = vn;
    __syncthreads();

    if (s + 1 < 4096) {  // prefetch next step
      long nb = base0 + (long)(s + 1) * 1536;
      qn = qkv[nb + dv]; kn = qkv[nb + 512 + dv]; vn = qkv[nb + 1024 + dv];
      bnx = beta[sb0 + (long)(s + 1) * 8];
      gnx = g[sb0 + (long)(s + 1) * 8];
    }

    const float bi = sc[0], gi = sc[1];
    float dot = 0.f;
    #pragma unroll
    for (int i = 0; i < 64; ++i) dot = fmaf(ks[i], S[i], dot);
    const float err = bi * (vv - gi * dot);   // pred = gi*dot (decay folded)
    float outv = 0.f;
    #pragma unroll
    for (int i = 0; i < 64; ++i) {
      S[i] = fmaf(gi, S[i], ks[i] * err);
      outv = fmaf(qs[i], S[i], outv);
    }
    o[ob0 + (long)s * 512 + dv] = outv;
  }
}

// ---------------------------------------------------------------------------
// silu(gate)*up -> f16, reading fused gate/up [R,4096] (gate:0, up:2048)
// ---------------------------------------------------------------------------
__global__ void silu_mul_kernel(const float* __restrict__ gu,
                                _Float16* __restrict__ out, long n) {
  long i = (long)blockIdx.x * blockDim.x + threadIdx.x;
  if (i < n) {
    long row = i >> 11;
    int  c   = (int)(i & 2047);
    float x = gu[row * 4096 + c];
    float u = gu[row * 4096 + 2048 + c];
    out[i] = (_Float16)(x / (1.f + expf(-x)) * u);
  }
}

__global__ void add_kernel(const float* __restrict__ a,
                           const float* __restrict__ b,
                           float* __restrict__ out, long n) {
  long i = (long)blockIdx.x * blockDim.x + threadIdx.x;
  if (i < n) out[i] = a[i] + b[i];
}

// ---------------------------------------------------------------------------
// Orchestration
// ---------------------------------------------------------------------------
extern "C" void kernel_launch(void* const* d_in, const int* in_sizes, int n_in,
                              void* d_out, int out_size, void* d_ws,
                              size_t ws_size, hipStream_t stream) {
  (void)in_sizes; (void)n_in; (void)out_size; (void)ws_size;
  const float* x     = (const float*)d_in[0];
  const float* mask  = (const float*)d_in[1];
  const float* n1w   = (const float*)d_in[2];
  const float* wq    = (const float*)d_in[3];
  const float* wk    = (const float*)d_in[4];
  const float* wv    = (const float*)d_in[5];
  const float* wb    = (const float*)d_in[6];
  const float* wg    = (const float*)d_in[7];
  const float* wo    = (const float*)d_in[8];
  const float* n2w   = (const float*)d_in[9];
  const float* wgate = (const float*)d_in[10];
  const float* wup   = (const float*)d_in[11];
  const float* wdown = (const float*)d_in[12];
  float* out = (float*)d_out;

  const long R = 8192;  // B*S

  char* p = (char*)d_ws;
  auto carve = [&](size_t bytes) -> char* {
    char* r = p;
    p += (bytes + 255) & ~(size_t)255;
    return r;
  };
  _Float16* hn    = (_Float16*)carve(R * 1024 * 2);     // rmsnorm1 out (f16)
  _Float16* wqkvT = (_Float16*)carve(1536 * 1024 * 2);  // [1536,1024] fused q|k|v ^T
  _Float16* woT   = (_Float16*)carve(1024 * 512 * 2);   // [1024,512]
  _Float16* wguT  = (_Float16*)carve(4096 * 1024 * 2);  // [4096,1024] fused gate|up ^T
  _Float16* wdT   = (_Float16*)carve(1024 * 2048 * 2);  // [1024,2048]
  float* qkv  = (float*)carve(R * 1536 * 4);            // fused q|k|v
  float* beta = (float*)carve(R * 8 * 4);
  float* gdec = (float*)carve(R * 8 * 4);
  float* ob   = (float*)carve(R * 512 * 4);
  _Float16* oh = (_Float16*)carve(R * 512 * 2);
  float* attn = (float*)carve(R * 1024 * 4);
  float* hres = (float*)carve(R * 1024 * 4);
  _Float16* h2 = (_Float16*)carve(R * 1024 * 2);
  float* gu   = (float*)carve(R * 4096 * 4);            // fused gate|up
  _Float16* act = (_Float16*)carve(R * 2048 * 2);
  float* down = (float*)carve(R * 1024 * 4);

  auto cvtT = [&](const float* src, _Float16* dst, int K, int N) {
    cvt_tr_kernel<<<dim3(N / 32, K / 32), dim3(32, 8), 0, stream>>>(src, dst, K, N);
  };
  auto gemm = [&](const _Float16* A, const _Float16* Bt, float* C, int M, int N, int K) {
    gemm_f16_wmma_kernel<<<dim3(N / 128, M / 128), 256, 0, stream>>>(A, Bt, C, M, N, K);
  };

  // Weights -> f16 transposed [N,K] (fused blocks are contiguous N-rows)
  cvtT(wq, wqkvT, 1024, 512);
  cvtT(wk, wqkvT + (size_t)512 * 1024, 1024, 512);
  cvtT(wv, wqkvT + (size_t)1024 * 1024, 1024, 512);
  cvtT(wo, woT, 512, 1024);
  cvtT(wgate, wguT, 1024, 2048);
  cvtT(wup, wguT + (size_t)2048 * 1024, 1024, 2048);
  cvtT(wdown, wdT, 2048, 1024);

  // 1) rmsnorm1 -> f16
  rmsnorm_kernel<<<dim3((unsigned)R), 256, 0, stream>>>(x, n1w, hn, 1024);

  // 2) fused QKV projection (one pass over hn)
  gemm(hn, wqkvT, qkv, (int)R, 1536, 1024);

  // 3) beta / decay gates
  betag_kernel<<<dim3((unsigned)R), 32, 0, stream>>>(hn, wb, wg, mask, beta, gdec);

  // 4) k L2-normalize + q scale (in place in fused qkv)
  prep_qk_kernel<<<dim3((unsigned)R), 256, 0, stream>>>(qkv);

  // 5) sequential delta-rule scan (16 chains)
  delta_scan_kernel<<<dim3(16), 64, 0, stream>>>(qkv, beta, gdec, ob);

  // 6) output projection + residual
  cvt_f16_kernel<<<dim3((unsigned)((R * 512 + 255) / 256)), 256, 0, stream>>>(
      ob, oh, R * 512);
  gemm(oh, woT, attn, (int)R, 1024, 512);
  add_kernel<<<dim3((unsigned)((R * 1024 + 255) / 256)), 256, 0, stream>>>(
      x, attn, hres, R * 1024);

  // 7) rmsnorm2 -> f16
  rmsnorm_kernel<<<dim3((unsigned)R), 256, 0, stream>>>(hres, n2w, h2, 1024);

  // 8) SwiGLU FFN: fused gate|up GEMM, silu*mul, down GEMM
  gemm(h2, wguT, gu, (int)R, 4096, 1024);
  silu_mul_kernel<<<dim3((unsigned)((R * 2048 + 255) / 256)), 256, 0, stream>>>(
      gu, act, R * 2048);
  gemm(act, wdT, down, (int)R, 1024, 2048);

  // 9) final residual -> d_out
  add_kernel<<<dim3((unsigned)((R * 1024 + 255) / 256)), 256, 0, stream>>>(
      hres, down, out, R * 1024);
}
